// AttentionModel_61675730371152
// MI455X (gfx1250) — matrix operand order repaired and checked by reading, not compiled, not run on hardware
//
#include <hip/hip_runtime.h>
#include <hip/hip_bf16.h>
#include <cstdint>
#include <cstddef>

#define S_LEN 512
#define B_SZ  128
#define I_SZ  128
#define H_SZ  256
#define BPB   16                 // batch rows per block
#define NBLK  (B_SZ / BPB)       // 8 persistent blocks
#define TPB   256                // 8 waves (wave32)
#define HPAD  (H_SZ + 8)         // padded bf16 row stride: 528 B = 33*16, keeps 16B alignment

typedef __bf16 v16bf __attribute__((ext_vector_type(16)));
typedef float  v8f   __attribute__((ext_vector_type(8)));
typedef unsigned v4u __attribute__((ext_vector_type(4)));   // native vector: OK for nontemporal builtins

union ABfrag { v16bf v; v4u q[2]; unsigned u[8]; };

// ---- bf16 helpers (round-to-nearest-even) ----
#if __has_builtin(__builtin_amdgcn_cvt_pk_bf16_f32)
typedef __bf16 v2bf __attribute__((ext_vector_type(2)));
__device__ __forceinline__ unsigned pack2_bf16(float lo, float hi) {
  union { v2bf v; unsigned u; } c;
  c.v = __builtin_amdgcn_cvt_pk_bf16_f32(lo, hi);
  return c.u;
}
#else
__device__ __forceinline__ unsigned pack2_bf16(float lo, float hi) {
  union { float f; unsigned u; } a, b;
  a.f = lo; b.f = hi;
  unsigned ua = a.u + (0x7FFFu + ((a.u >> 16) & 1u));
  unsigned ub = b.u + (0x7FFFu + ((b.u >> 16) & 1u));
  return (ua >> 16) | (ub & 0xFFFF0000u);
}
#endif
__device__ __forceinline__ unsigned short f2bf(float x) {
  return (unsigned short)(pack2_bf16(x, 0.f) & 0xFFFFu);
}
__device__ __forceinline__ float bf2f(unsigned short h) {
  union { unsigned u; float f; } a; a.u = ((unsigned)h) << 16;
  return a.f;
}
__device__ __forceinline__ float2 bf2x2(unsigned u) {
  union { unsigned v; float f; } lo, hi;
  lo.v = u << 16; hi.v = u & 0xFFFF0000u;
  return make_float2(lo.f, hi.f);
}

// B fragment (32x16 bf16): per lane, two 16B runs at row n, elements
// kc*32 + 8*hi and kc*32 + 16 + 8*hi  ->  two b128 loads.
__device__ __forceinline__ void load_bfrag(ABfrag& f, const unsigned* w,
                                           int n, int K, int kc, int hi) {
  const unsigned* p = w + ((n * K) >> 1) + kc * 16 + hi * 4;
  f.q[0] = *(const v4u*)p;
  f.q[1] = *(const v4u*)(p + 8);
}
// A fragment (16x32 bf16) from LDS: two 16B runs at elements
// base, base+16 where base = row*HPAD + kc*32 + 8*hi (all 16B aligned).
__device__ __forceinline__ void load_afrag_lds(ABfrag& f, const unsigned short* buf,
                                               int row, int kc, int hi) {
  const unsigned* p = (const unsigned*)&buf[row * HPAD + kc * 32 + 8 * hi];
  f.q[0] = *(const v4u*)p;
  f.q[1] = *(const v4u*)(p + 8);
}

// ------------------------------------------------------------------
// fp32 -> bf16 conversion pre-pass (weights + X). Moves all fp32->bf16
// packing out of the sequential recurrences into a parallel, HBM-bound
// kernel (~50 MB total traffic ≈ 2 us at 23.3 TB/s).
// ------------------------------------------------------------------
__global__ void cvt_bf16_kernel(const float* __restrict__ in,
                                unsigned short* __restrict__ outp, int n) {
  int i = blockIdx.x * blockDim.x + threadIdx.x;
  if (i < n) outp[i] = f2bf(in[i]);
}

// ------------------------------------------------------------------
// Fused input-projection + encoder RNN scan. One block per 16 batch
// rows; hidden tile (16xH, bf16) lives in LDS across the 512 steps.
// ------------------------------------------------------------------
__global__ __launch_bounds__(TPB) void encoder_kernel(
    const unsigned* __restrict__ xb,         // bf16 [S,B,I] (u32 = 2 bf16)
    const unsigned* __restrict__ wih,        // bf16 [H,I]
    const unsigned* __restrict__ whh,        // bf16 [H,H]
    const float* __restrict__ b_ih,
    const float* __restrict__ b_hh,
    unsigned short* __restrict__ out_pre)    // bf16 [S,B,H]
{
  __shared__ unsigned short h_lds[BPB * HPAD];

  const int tid  = threadIdx.x;
  const int lane = tid & 31;
  const int wave = tid >> 5;
  const int hi   = (lane >> 4) & 1;
  const int ln   = lane & 15;
  const int b0   = blockIdx.x * BPB;
  const int n0   = wave * 32 + ln;           // first output-column tile
  const int n1   = n0 + 16;                  // second tile

  const float bias0 = b_ih[n0] + b_hh[n0];
  const float bias1 = b_ih[n1] + b_hh[n1];

  for (int i = tid; i < BPB * HPAD; i += TPB) h_lds[i] = 0;
  __syncthreads();

  for (int s = 0; s < S_LEN; ++s) {
    v8f acc0, acc1;
#pragma unroll
    for (int r = 0; r < 8; ++r) { acc0[r] = bias0; acc1[r] = bias1; }

    // ---- input projection: Xb[s, b0+ln, :] (16x128) @ W_ih^T ----
    const unsigned* xrow = xb + (((size_t)s * B_SZ + (b0 + ln)) * I_SZ >> 1);
#pragma unroll 2
    for (int kc = 0; kc < I_SZ / 32; ++kc) {
      ABfrag a, bb0, bb1;
      const unsigned* pa = xrow + kc * 16 + hi * 4;
      a.q[0] = __builtin_nontemporal_load((const v4u*)pa);        // X streamed once
      a.q[1] = __builtin_nontemporal_load((const v4u*)(pa + 8));
      load_bfrag(bb0, wih, n0, I_SZ, kc, hi);
      load_bfrag(bb1, wih, n1, I_SZ, kc, hi);
      acc0 = __builtin_amdgcn_wmma_f32_16x16x32_bf16(false, a.v, false, bb0.v, (short)0, acc0, false, false);
      acc1 = __builtin_amdgcn_wmma_f32_16x16x32_bf16(false, a.v, false, bb1.v, (short)0, acc1, false, false);
    }

    // ---- recurrent: h (16x256) @ W_hh^T  (h == 0 at s == 0) ----
    if (s > 0) {
#pragma unroll 2
      for (int kc = 0; kc < H_SZ / 32; ++kc) {
        ABfrag a, bb0, bb1;
        load_afrag_lds(a, h_lds, ln, kc, hi);
        load_bfrag(bb0, whh, n0, H_SZ, kc, hi);
        load_bfrag(bb1, whh, n1, H_SZ, kc, hi);
        acc0 = __builtin_amdgcn_wmma_f32_16x16x32_bf16(false, a.v, false, bb0.v, (short)0, acc0, false, false);
        acc1 = __builtin_amdgcn_wmma_f32_16x16x32_bf16(false, a.v, false, bb1.v, (short)0, acc1, false, false);
      }
    }
    __syncthreads();   // all waves done reading previous h

#pragma unroll
    for (int r = 0; r < 8; ++r) {
      int m = r + 8 * hi;                    // C/D layout: VGPR r -> M = r + 8*hi
      unsigned short v0 = f2bf(tanhf(acc0[r]));
      unsigned short v1 = f2bf(tanhf(acc1[r]));
      h_lds[m * HPAD + n0] = v0;
      h_lds[m * HPAD + n1] = v1;
      size_t o = ((size_t)s * B_SZ + (b0 + m)) * H_SZ;
      out_pre[o + n0] = v0;
      out_pre[o + n1] = v1;
    }
    __syncthreads();   // new h visible for next step
  }
}

// ------------------------------------------------------------------
// Persistent attention recursion: 256 iterations, one block per 16
// batch rows (each row's softmax/attention/cell update is independent).
// ------------------------------------------------------------------
__global__ __launch_bounds__(TPB) void attention_kernel(
    const unsigned short* __restrict__ out_pre, // bf16 [S,B,H]
    const unsigned* __restrict__ wihc,          // bf16 [H,H]
    const unsigned* __restrict__ whhc,          // bf16 [H,H]
    const float* __restrict__ b_ihc,
    const float* __restrict__ b_hhc,
    const float* __restrict__ fc_w,             // [1,H]
    const float* __restrict__ fc_b,             // [1]
    float* __restrict__ out)                    // [B,1] fp32
{
  __shared__ float ps[S_LEN][BPB + 1];          // scores / softmax weights
  __shared__ unsigned short hp_lds[BPB * HPAD]; // hidden_post (bf16)
  __shared__ unsigned short attn_lds[BPB * HPAD];
  __shared__ float red[BPB * 16];

  const int tid  = threadIdx.x;
  const int lane = tid & 31;
  const int wave = tid >> 5;
  const int hi   = (lane >> 4) & 1;
  const int ln   = lane & 15;
  const int b0   = blockIdx.x * BPB;
  const int n0   = wave * 32 + ln;
  const int n1   = n0 + 16;
  const float bias0 = b_ihc[n0] + b_hhc[n0];
  const float bias1 = b_ihc[n1] + b_hhc[n1];

  const int j_a = tid & 15;    // batch row for phases a/c
  const int sg  = tid >> 4;    // 0..15

  for (int i = tid; i < BPB * HPAD; i += TPB) { hp_lds[i] = 0; attn_lds[i] = 0; }
  __syncthreads();

  for (int it = 0; it < H_SZ; ++it) {
    // -- phase a: ps[s][j] = <out_pre[s, b0+j, :], hp[j, :]> --
    const v4u* hrow = (const v4u*)&hp_lds[j_a * HPAD];   // 528B stride, 16B aligned
    for (int ss = 0; ss < 32; ++ss) {
      int s = sg * 32 + ss;
      const v4u* orow =
          (const v4u*)(out_pre + ((size_t)s * B_SZ + (b0 + j_a)) * H_SZ);
      float acc = 0.f;
#pragma unroll 4
      for (int c = 0; c < H_SZ / 8; ++c) {
        v4u o = __builtin_nontemporal_load(orow + c);
        v4u h4 = hrow[c];
        float2 o0 = bf2x2(o.x), o1 = bf2x2(o.y), o2 = bf2x2(o.z), o3 = bf2x2(o.w);
        float2 h0 = bf2x2(h4.x), h1 = bf2x2(h4.y), h2 = bf2x2(h4.z), h3 = bf2x2(h4.w);
        acc += o0.x * h0.x + o0.y * h0.y + o1.x * h1.x + o1.y * h1.y +
               o2.x * h2.x + o2.y * h2.y + o3.x * h3.x + o3.y * h3.y;
      }
      ps[s][j_a] = acc;
    }
    __syncthreads();

    // -- phase b: softmax over s (wave w handles rows 2w, 2w+1) --
#pragma unroll
    for (int jj = 0; jj < 2; ++jj) {
      int j = wave * 2 + jj;
      float vals[S_LEN / 32];
      float m = -3.4e38f;
#pragma unroll
      for (int k = 0; k < S_LEN / 32; ++k) {
        vals[k] = ps[k * 32 + lane][j];
        m = fmaxf(m, vals[k]);
      }
#pragma unroll
      for (int off = 16; off > 0; off >>= 1) m = fmaxf(m, __shfl_xor(m, off, 32));
      float sum = 0.f;
#pragma unroll
      for (int k = 0; k < S_LEN / 32; ++k) { vals[k] = __expf(vals[k] - m); sum += vals[k]; }
#pragma unroll
      for (int off = 16; off > 0; off >>= 1) sum += __shfl_xor(sum, off, 32);
      float inv = 1.f / sum;
#pragma unroll
      for (int k = 0; k < S_LEN / 32; ++k) ps[k * 32 + lane][j] = vals[k] * inv;
    }
    __syncthreads();

    // -- phase c: attn[j, 16hc..16hc+15] = sum_s ps[s][j] * out_pre[s, b0+j, :] --
    {
      const int hc = sg;
      float acc[16];
#pragma unroll
      for (int k = 0; k < 16; ++k) acc[k] = 0.f;
      for (int s = 0; s < S_LEN; ++s) {
        float w = ps[s][j_a];
        const v4u* orow =
            (const v4u*)(out_pre + ((size_t)s * B_SZ + (b0 + j_a)) * H_SZ + hc * 16);
        v4u oA = __builtin_nontemporal_load(orow);
        v4u oB = __builtin_nontemporal_load(orow + 1);
        float2 f;
        f = bf2x2(oA.x); acc[0] += w * f.x;  acc[1] += w * f.y;
        f = bf2x2(oA.y); acc[2] += w * f.x;  acc[3] += w * f.y;
        f = bf2x2(oA.z); acc[4] += w * f.x;  acc[5] += w * f.y;
        f = bf2x2(oA.w); acc[6] += w * f.x;  acc[7] += w * f.y;
        f = bf2x2(oB.x); acc[8] += w * f.x;  acc[9] += w * f.y;
        f = bf2x2(oB.y); acc[10] += w * f.x; acc[11] += w * f.y;
        f = bf2x2(oB.z); acc[12] += w * f.x; acc[13] += w * f.y;
        f = bf2x2(oB.w); acc[14] += w * f.x; acc[15] += w * f.y;
      }
      v4u wA, wB;
      wA.x = pack2_bf16(acc[0], acc[1]);   wA.y = pack2_bf16(acc[2], acc[3]);
      wA.z = pack2_bf16(acc[4], acc[5]);   wA.w = pack2_bf16(acc[6], acc[7]);
      wB.x = pack2_bf16(acc[8], acc[9]);   wB.y = pack2_bf16(acc[10], acc[11]);
      wB.z = pack2_bf16(acc[12], acc[13]); wB.w = pack2_bf16(acc[14], acc[15]);
      v4u* arow = (v4u*)&attn_lds[j_a * HPAD + hc * 16];   // 16B aligned
      arow[0] = wA;
      arow[1] = wB;
    }
    __syncthreads();

    // -- phase d: hp = tanh(hp@Wih_c^T + attn@Whh_c^T + bias)  (WMMA) --
    v8f acc0, acc1;
#pragma unroll
    for (int r = 0; r < 8; ++r) { acc0[r] = bias0; acc1[r] = bias1; }
#pragma unroll 2
    for (int kc = 0; kc < H_SZ / 32; ++kc) {
      ABfrag ah, aa, b0h, b1h, b0a, b1a;
      load_afrag_lds(ah, hp_lds, ln, kc, hi);
      load_afrag_lds(aa, attn_lds, ln, kc, hi);
      load_bfrag(b0h, wihc, n0, H_SZ, kc, hi);
      load_bfrag(b1h, wihc, n1, H_SZ, kc, hi);
      load_bfrag(b0a, whhc, n0, H_SZ, kc, hi);
      load_bfrag(b1a, whhc, n1, H_SZ, kc, hi);
      acc0 = __builtin_amdgcn_wmma_f32_16x16x32_bf16(false, ah.v, false, b0h.v, (short)0, acc0, false, false);
      acc1 = __builtin_amdgcn_wmma_f32_16x16x32_bf16(false, ah.v, false, b1h.v, (short)0, acc1, false, false);
      acc0 = __builtin_amdgcn_wmma_f32_16x16x32_bf16(false, aa.v, false, b0a.v, (short)0, acc0, false, false);
      acc1 = __builtin_amdgcn_wmma_f32_16x16x32_bf16(false, aa.v, false, b1a.v, (short)0, acc1, false, false);
    }
    __syncthreads();   // everyone finished reading hp_lds
#pragma unroll
    for (int r = 0; r < 8; ++r) {
      int m = r + 8 * hi;
      hp_lds[m * HPAD + n0] = f2bf(tanhf(acc0[r]));
      hp_lds[m * HPAD + n1] = f2bf(tanhf(acc1[r]));
    }
    __syncthreads();
  }

  // -- final FC: out[b0+j] = <hp[j,:], fc_w> + fc_b --
  {
    float part = 0.f;
#pragma unroll
    for (int k = 0; k < 16; ++k) {
      int h = sg * 16 + k;
      part += bf2f(hp_lds[j_a * HPAD + h]) * fc_w[h];
    }
    red[sg * 16 + j_a] = part;
    __syncthreads();
    if (tid < BPB) {
      float s = fc_b[0];
#pragma unroll
      for (int g = 0; g < 16; ++g) s += red[g * 16 + tid];
      out[b0 + tid] = s;
    }
  }
}

extern "C" void kernel_launch(void* const* d_in, const int* in_sizes, int n_in,
                              void* d_out, int out_size, void* d_ws, size_t ws_size,
                              hipStream_t stream) {
  const float* X        = (const float*)d_in[0];
  const float* W_ih_pre = (const float*)d_in[1];
  const float* W_hh_pre = (const float*)d_in[2];
  const float* b_ih_pre = (const float*)d_in[3];
  const float* b_hh_pre = (const float*)d_in[4];
  const float* W_ih_c   = (const float*)d_in[5];
  const float* W_hh_c   = (const float*)d_in[6];
  const float* b_ih_c   = (const float*)d_in[7];
  const float* b_hh_c   = (const float*)d_in[8];
  const float* fc_w     = (const float*)d_in[9];
  const float* fc_b     = (const float*)d_in[10];

  // workspace layout (bytes)
  constexpr size_t OUTPRE_B = (size_t)S_LEN * B_SZ * H_SZ * 2;   // 32 MB  bf16 out_pre
  constexpr size_t XB_B     = (size_t)S_LEN * B_SZ * I_SZ * 2;   // 16 MB  bf16 X
  constexpr int    NX       = S_LEN * B_SZ * I_SZ;               // 8.4M elements
  char* ws = (char*)d_ws;
  unsigned short* out_pre = (unsigned short*)ws;
  unsigned short* xbw  = (unsigned short*)(ws + OUTPRE_B);
  unsigned short* wih  = (unsigned short*)(ws + OUTPRE_B + XB_B);
  unsigned short* whh  = (unsigned short*)(ws + OUTPRE_B + XB_B + 65536);
  unsigned short* wihc = (unsigned short*)(ws + OUTPRE_B + XB_B + 65536 + 131072);
  unsigned short* whhc = (unsigned short*)(ws + OUTPRE_B + XB_B + 65536 + 2 * 131072);

  cvt_bf16_kernel<<<(NX + 255) / 256, 256, 0, stream>>>(X, xbw, NX);
  cvt_bf16_kernel<<<(H_SZ * I_SZ + 255) / 256, 256, 0, stream>>>(W_ih_pre, wih, H_SZ * I_SZ);
  cvt_bf16_kernel<<<(H_SZ * H_SZ + 255) / 256, 256, 0, stream>>>(W_hh_pre, whh, H_SZ * H_SZ);
  cvt_bf16_kernel<<<(H_SZ * H_SZ + 255) / 256, 256, 0, stream>>>(W_ih_c, wihc, H_SZ * H_SZ);
  cvt_bf16_kernel<<<(H_SZ * H_SZ + 255) / 256, 256, 0, stream>>>(W_hh_c, whhc, H_SZ * H_SZ);

  encoder_kernel<<<NBLK, TPB, 0, stream>>>(
      (const unsigned*)xbw, (const unsigned*)wih, (const unsigned*)whh,
      b_ih_pre, b_hh_pre, out_pre);

  attention_kernel<<<NBLK, TPB, 0, stream>>>(
      out_pre, (const unsigned*)wihc, (const unsigned*)whhc,
      b_ih_c, b_hh_c, fc_w, fc_b, (float*)d_out);
}
